// LstmWeightDrop_63428077027869
// MI455X (gfx1250) — matrix operand order repaired and checked by reading, not compiled, not run on hardware
//
#include <hip/hip_runtime.h>

// ---------------------------------------------------------------------------
// MI455X (gfx1250) 2-layer bidirectional LSTM
//   B=64, T=1024, I=256, H=256, gates order i,f,g,o (PyTorch)
//   - input GEMMs hoisted into big parallel v_wmma_f32_16x16x32_f16 GEMMs,
//     tiled as (16 batches x 16 gate-cols) at fixed t so the C tiles match
//     the scan's gate tiles exactly
//   - xg stored in WMMA C register layout [dir][t][nt][mt][lane][8]:
//     GEMM stores and scan gate-init loads are contiguous b128 ops
//   - recurrence: persistent workgroup per direction, h kept in LDS (f16,
//     double-buffered), c in registers, 1 barrier per timestep
// ---------------------------------------------------------------------------

typedef __attribute__((ext_vector_type(16))) _Float16     v16h;
typedef __attribute__((ext_vector_type(8)))  float        v8f;
typedef __attribute__((ext_vector_type(4)))  float        v4f;
typedef __attribute__((ext_vector_type(4)))  unsigned int v4u;

union H16 { v16h v; v4u q[2]; };
union F8  { v8f  v; v4f q[2]; };

#define Tn 1024
#define Bn 64
#define Hn 256
#define G4 1024   // 4*H

__device__ __forceinline__ float sigf(float x) { return 1.0f / (1.0f + __expf(-x)); }

// --------------------------- f32 -> f16 convert ----------------------------
__global__ void cvt_f16_kernel(const float* __restrict__ in,
                               _Float16* __restrict__ out, int n) {
  int i = blockIdx.x * blockDim.x + threadIdx.x;
  int stride = gridDim.x * blockDim.x;
  for (; i < n; i += stride) out[i] = (_Float16)in[i];
}

// --------------------------- hoisted input GEMM ----------------------------
// For fixed t: G[b][N] = x[b][t][:] @ W[dir][N][:] + b_ih + b_hh
// A: [B][T][K] f16 (row (b,t) contiguous in K), W: [2][4H][K] f16 row-major.
// Tile: 16 batches x 16 gate-cols per wave. Output stored in WMMA C register
// layout: xg[dir][t][nt(64)][mt(4)][lane(32)][8] -> contiguous v8f per lane.
__global__ __launch_bounds__(256)
void gemm_xg_kernel(const _Float16* __restrict__ A,
                    const _Float16* __restrict__ W,
                    const float* __restrict__ b_ih,
                    const float* __restrict__ b_hh,
                    float* __restrict__ xg, int K) {
  const int lane = threadIdx.x & 31;
  const int wave = threadIdx.x >> 5;
  const int t    = blockIdx.x >> 2;          // timestep
  const int mt   = blockIdx.x & 3;           // batch tile (0..3)
  const int nt   = blockIdx.y * 8 + wave;    // gate-col tile (0..63)
  const int dir  = blockIdx.z;
  const int ncol = lane & 15;
  const int hi   = (lane >> 4) & 1;
  const int N    = nt * 16 + ncol;

  const float bias = b_ih[dir * G4 + N] + b_hh[dir * G4 + N];
  v8f acc;
#pragma unroll
  for (int r = 0; r < 8; ++r) acc[r] = bias;

  // A lane layout (16x32 f16): two contiguous 8-half runs at kb+{0,16}(+8 if hi)
  // B lane layout (32x16 f16): two contiguous 8-half runs at kb+{0,16}(+8 if hi), row = N
  const _Float16* Arow = A + ((size_t)(mt * 16 + ncol) * Tn + t) * K;
  const _Float16* Wrow = W + ((size_t)dir * G4 + N) * K;
  const int aoff = hi * 8;
  const int boff = hi * 8;

  for (int kb = 0; kb < K; kb += 32) {
    H16 a, b;
    a.q[0] = *(const v4u*)(Arow + kb + aoff);
    a.q[1] = *(const v4u*)(Arow + kb + aoff + 16);
    b.q[0] = *(const v4u*)(Wrow + kb + boff);
    b.q[1] = *(const v4u*)(Wrow + kb + boff + 16);
    acc = __builtin_amdgcn_wmma_f32_16x16x32_f16(false, a.v, false, b.v,
                                                 (short)0, acc, false, false);
  }

  // store acc in C register layout: wave writes one contiguous 1KB blob
  float* dst = xg + ((((size_t)(dir * Tn + t) * 64 + nt) * 4 + mt) * 256) + lane * 8;
  F8 o; o.v = acc;
  *(v4f*)(dst)     = o.q[0];
  *(v4f*)(dst + 4) = o.q[1];
}

// ------------------------------ LSTM scan ----------------------------------
// grid = 2 blocks (one per direction), 1024 threads = 32 waves.
// Wave w: batch tile m = w&3, j-tiles jt0=(w>>2)*2 .. +1; for each unit it
// computes ALL FOUR gate tiles (same j columns) so the c/h update is local.
__global__ __launch_bounds__(1024)
void lstm_scan_kernel(const float* __restrict__ xg,       // [2][t][nt][mt][256]
                      const _Float16* __restrict__ whh,   // [2][4H][H]
                      _Float16* __restrict__ out_h,       // layer0: o1 f16 [B][T][2H]
                      float* __restrict__ out_f,          // layer1: o2 f32 [B][T][2H]
                      float* __restrict__ hn,             // [4][B][H]
                      float* __restrict__ cn,             // [4][B][H]
                      int slot) {
  __shared__ _Float16 hbuf[2 * Bn * Hn];   // double-buffered h (64KB)

  const int dir  = blockIdx.x;
  const int tid  = threadIdx.x;
  const int lane = tid & 31;
  const int wave = tid >> 5;
  const int m    = wave & 3;
  const int jt0  = (wave >> 2) << 1;
  const int ncol = lane & 15;
  const int hi   = (lane >> 4) & 1;

  for (int i = tid; i < 2 * Bn * Hn; i += 1024) hbuf[i] = (_Float16)0.0f;

  float c[2][8];
  float h[2][8];
#pragma unroll
  for (int u = 0; u < 2; ++u)
#pragma unroll
    for (int r = 0; r < 8; ++r) { c[u][r] = 0.0f; h[u][r] = 0.0f; }

  const float*    xg_d = xg  + (size_t)dir * Tn * Bn * G4;
  const _Float16* w_d  = whh + (size_t)dir * G4 * Hn;

  for (int s = 0; s < Tn; ++s) {
    __syncthreads();  // prev-step h writes visible; nobody still reads old buf
    const int t = dir ? (Tn - 1 - s) : s;
    const _Float16* cur = hbuf + (s & 1) * (Bn * Hn);
    _Float16*       nxt = hbuf + ((s + 1) & 1) * (Bn * Hn);
    const float*   xg_t = xg_d + (size_t)t * (Bn * G4);

    if (s + 1 < Tn) {  // prefetch next timestep's gate slab (256KB, 1 line/thread)
      const int tnext = dir ? (Tn - 2 - s) : (s + 1);
      __builtin_prefetch(xg_d + (size_t)tnext * (Bn * G4) + tid * 64, 0, 0);
    }

    // gate init: xg already in WMMA C register layout -> contiguous b128 loads
    v8f acc[2][4];
#pragma unroll
    for (int u = 0; u < 2; ++u) {
      const int jt = jt0 + u;
#pragma unroll
      for (int g = 0; g < 4; ++g) {
        const int nt = g * 16 + jt;
        const float* src = xg_t + ((size_t)(nt * 4 + m) * 256) + lane * 8;
        F8 in;
        in.q[0] = *(const v4f*)(src);
        in.q[1] = *(const v4f*)(src + 4);
        acc[u][g] = in.v;
      }
    }

    // g += h @ Whh^T : K-loop over H in 32-chunks, A from LDS, B from L2
    const _Float16* arow = cur + (m * 16 + ncol) * Hn;
    for (int kb = 0; kb < Hn; kb += 32) {
      H16 a;
      a.q[0] = *(const v4u*)(arow + kb + hi * 8);
      a.q[1] = *(const v4u*)(arow + kb + hi * 8 + 16);
#pragma unroll
      for (int u = 0; u < 2; ++u) {
        const int jt = jt0 + u;
#pragma unroll
        for (int g = 0; g < 4; ++g) {
          const int N = g * Hn + jt * 16 + ncol;
          const _Float16* brow = w_d + (size_t)N * Hn + kb + hi * 8;
          H16 b;
          b.q[0] = *(const v4u*)(brow);
          b.q[1] = *(const v4u*)(brow + 16);
          acc[u][g] = __builtin_amdgcn_wmma_f32_16x16x32_f16(
              false, a.v, false, b.v, (short)0, acc[u][g], false, false);
        }
      }
    }

    // elementwise LSTM cell update (all gates are wave-local)
#pragma unroll
    for (int u = 0; u < 2; ++u) {
      const int jt = jt0 + u;
#pragma unroll
      for (int r = 0; r < 8; ++r) {
        const int bl = m * 16 + hi * 8 + r;     // batch index
        const int j  = jt * 16 + ncol;          // hidden index
        const float iv = sigf(acc[u][0][r]);
        const float fv = sigf(acc[u][1][r]);
        const float gv = tanhf(acc[u][2][r]);
        const float ov = sigf(acc[u][3][r]);
        const float cv = fv * c[u][r] + iv * gv;
        const float hv = ov * tanhf(cv);
        c[u][r] = cv;
        h[u][r] = hv;
        nxt[bl * Hn + j] = (_Float16)hv;
        const size_t oidx = ((size_t)bl * Tn + t) * (2 * Hn) + (size_t)dir * Hn + j;
        if (out_h) out_h[oidx] = (_Float16)hv;
        else       out_f[oidx] = hv;
      }
    }
  }

  // final h_n / c_n
#pragma unroll
  for (int u = 0; u < 2; ++u) {
    const int jt = jt0 + u;
#pragma unroll
    for (int r = 0; r < 8; ++r) {
      const int bl = m * 16 + hi * 8 + r;
      const int j  = jt * 16 + ncol;
      const size_t idx = ((size_t)(slot + dir) * Bn + bl) * Hn + j;
      hn[idx] = h[u][r];
      cn[idx] = c[u][r];
    }
  }
}

// ------------------------------- launcher ----------------------------------
extern "C" void kernel_launch(void* const* d_in, const int* in_sizes, int n_in,
                              void* d_out, int out_size, void* d_ws, size_t ws_size,
                              hipStream_t stream) {
  (void)in_sizes; (void)n_in; (void)out_size; (void)ws_size;

  const float* x    = (const float*)d_in[0];
  // d_in[1] = hidden_state (ignored by reference)
  const float* wih0 = (const float*)d_in[2];
  const float* whh0 = (const float*)d_in[3];
  const float* bih0 = (const float*)d_in[4];
  const float* bhh0 = (const float*)d_in[5];
  const float* wih1 = (const float*)d_in[6];
  const float* whh1 = (const float*)d_in[7];
  const float* bih1 = (const float*)d_in[8];
  const float* bhh1 = (const float*)d_in[9];

  char* ws = (char*)d_ws;
  size_t off = 0;
  _Float16* x_h    = (_Float16*)(ws + off); off += (size_t)Bn * Tn * 256 * 2;  //  32MB
  _Float16* o1_h   = (_Float16*)(ws + off); off += (size_t)Bn * Tn * 512 * 2;  //  64MB
  _Float16* wih0_h = (_Float16*)(ws + off); off += (size_t)2 * G4 * 256 * 2;   //   1MB
  _Float16* whh0_h = (_Float16*)(ws + off); off += (size_t)2 * G4 * 256 * 2;   //   1MB
  _Float16* wih1_h = (_Float16*)(ws + off); off += (size_t)2 * G4 * 512 * 2;   //   2MB
  _Float16* whh1_h = (_Float16*)(ws + off); off += (size_t)2 * G4 * 256 * 2;   //   1MB
  float*    xg     = (float*)(ws + off);    off += (size_t)2 * Tn * Bn * G4 * 4; // 512MB (reused layer0/1)

  float* out = (float*)d_out;
  float* o2  = out;                                   // [B][T][2H] f32
  float* hn  = out + (size_t)Bn * Tn * 2 * Hn;        // [4][B][H]
  float* cn  = hn + (size_t)4 * Bn * Hn;              // [4][B][H]

  // f32 -> f16 conversions
  cvt_f16_kernel<<<2048, 256, 0, stream>>>(x,    x_h,    Bn * Tn * 256);
  cvt_f16_kernel<<<512,  256, 0, stream>>>(wih0, wih0_h, 2 * G4 * 256);
  cvt_f16_kernel<<<512,  256, 0, stream>>>(whh0, whh0_h, 2 * G4 * 256);
  cvt_f16_kernel<<<512,  256, 0, stream>>>(wih1, wih1_h, 2 * G4 * 512);
  cvt_f16_kernel<<<512,  256, 0, stream>>>(whh1, whh1_h, 2 * G4 * 256);

  // layer 0: hoisted input GEMM + bidirectional scan (writes o1 f16)
  gemm_xg_kernel<<<dim3(4096, 8, 2), 256, 0, stream>>>(x_h, wih0_h, bih0, bhh0, xg, 256);
  lstm_scan_kernel<<<dim3(2), 1024, 0, stream>>>(xg, whh0_h, o1_h, nullptr, hn, cn, 0);

  // layer 1: input is o1 (2H=512 features), writes o2 f32 into d_out
  gemm_xg_kernel<<<dim3(4096, 8, 2), 256, 0, stream>>>(o1_h, wih1_h, bih1, bhh1, xg, 512);
  lstm_scan_kernel<<<dim3(2), 1024, 0, stream>>>(xg, whh1_h, nullptr, o2, hn, cn, 2);
}
